// MultiHeadSelfAttention_11751030522002
// MI455X (gfx1250) — matrix operand order repaired and checked by reading, not compiled, not run on hardware
//
#include <hip/hip_runtime.h>

typedef __attribute__((ext_vector_type(16))) __bf16 v16bf;
typedef __attribute__((ext_vector_type(16))) short  v16s;
typedef __attribute__((ext_vector_type(8)))  short  v8s;
typedef __attribute__((ext_vector_type(8)))  float  v8f;
typedef __attribute__((ext_vector_type(4)))  int    v4i;
typedef __attribute__((address_space(1))) v4i* as1_v4i;
typedef __attribute__((address_space(3))) v4i* as3_v4i;

#define DI __device__ __forceinline__

#if __has_builtin(__builtin_amdgcn_global_load_async_to_lds_b128)
#define HAVE_ASYNC_LDS 1
#else
#define HAVE_ASYNC_LDS 0
#endif

DI unsigned short f2bf(float f) {
  unsigned int u = __builtin_bit_cast(unsigned int, f);
  u += 0x7FFFu + ((u >> 16) & 1u);   // round-to-nearest-even
  return (unsigned short)(u >> 16);
}

DI v8f wmma_bf16(v16s a, v16s b, v8f c) {
  return __builtin_amdgcn_wmma_f32_16x16x32_bf16(
      false, __builtin_bit_cast(v16bf, a),
      false, __builtin_bit_cast(v16bf, b),
      (short)0, c, false, false);
}

// A fragment (16x32 bf16, one row per lane%16):
// lanes 0-15 hold K=[0,16), lanes 16-31 hold K=[16,32)
DI v16s load_a_frag(const unsigned short* rowp, int hf) {
  return *(const v16s*)(rowp + hf * 16);
}

// B fragment (32x16 bf16, one column per lane%16): 16 contiguous K at hf*16
DI v16s load_b_frag(const unsigned short* colp, int hf) {
  return *(const v16s*)(colp + hf * 16);
}

#if HAVE_ASYNC_LDS
DI void wait_async0() {
#if __has_builtin(__builtin_amdgcn_s_wait_asynccnt)
  __builtin_amdgcn_s_wait_asynccnt(0);
#else
  asm volatile("s_wait_asynccnt 0x0" ::: "memory");
#endif
}
// async copy of one thread's 32B slice (2 x b128), global -> LDS, no VGPR round-trip
DI void async_tile32(const unsigned short* g, unsigned short* l) {
  __builtin_amdgcn_global_load_async_to_lds_b128(
      (as1_v4i)(void*)g, (as3_v4i)(void*)l, 0, 0);
  __builtin_amdgcn_global_load_async_to_lds_b128(
      (as1_v4i)(void*)(g + 8), (as3_v4i)(void*)(l + 8), 0, 0);
}
#endif

__global__ __launch_bounds__(256)
void cvt_f32_bf16(const float* __restrict__ in, unsigned short* __restrict__ out, int n) {
  int i = blockIdx.x * 256 + threadIdx.x;
  if (i < n) out[i] = f2bf(in[i]);
}

// C[M,N] (f32, row-major) = A[M,K] (bf16 rm) x B[N,K]^T (bf16 rm)
// 128x128 block tile, K-tile 32, double-buffered LDS fed by async global->LDS.
__global__ __launch_bounds__(256)
void gemm_nt_bf16(const unsigned short* __restrict__ A,
                  const unsigned short* __restrict__ B,
                  float* __restrict__ C, int M, int N, int K) {
  __shared__ alignas(32) unsigned short As[2][128 * 32];
  __shared__ alignas(32) unsigned short Bs[2][128 * 32];
  const int tid  = threadIdx.x;
  const int lane = tid & 31, wid = tid >> 5;
  const int wm = wid >> 2, wn = wid & 3;              // 2 x 4 wave grid
  const int bm = blockIdx.y * 128, bn = blockIdx.x * 128;
  const int l16 = lane & 15, hf = lane >> 4;
  const int ldr = tid >> 1;                           // tile row this thread loads
  const int ldc = (tid & 1) * 16;                     // tile col half

  const unsigned short* gA = &A[(size_t)(bm + ldr) * K + ldc];
  const unsigned short* gB = &B[(size_t)(bn + ldr) * K + ldc];

  const v8f vz = {0, 0, 0, 0, 0, 0, 0, 0};
  v8f acc[4][2];
#pragma unroll
  for (int i = 0; i < 4; ++i)
#pragma unroll
    for (int j = 0; j < 2; ++j) acc[i][j] = vz;

  const int ntiles = K >> 5;

#if HAVE_ASYNC_LDS
  async_tile32(gA, &As[0][ldr * 32 + ldc]);
  async_tile32(gB, &Bs[0][ldr * 32 + ldc]);
  for (int kt = 0; kt < ntiles; ++kt) {
    const int cur = kt & 1, nxt = cur ^ 1;
    wait_async0();            // this thread's async fills for buffer `cur` done
    __syncthreads();          // everyone's fills visible; prior reads of `nxt` done
    if (kt + 1 < ntiles) {    // overlap next tile's DMA with this tile's WMMAs
      async_tile32(gA + (size_t)(kt + 1) * 32, &As[nxt][ldr * 32 + ldc]);
      async_tile32(gB + (size_t)(kt + 1) * 32, &Bs[nxt][ldr * 32 + ldc]);
    }
    v16s af[4], bfr[2];
#pragma unroll
    for (int i = 0; i < 4; ++i)
      af[i] = load_a_frag(&As[cur][(wm * 64 + i * 16 + l16) * 32], hf);
#pragma unroll
    for (int j = 0; j < 2; ++j)
      bfr[j] = load_b_frag(&Bs[cur][(wn * 32 + j * 16 + l16) * 32], hf);
#pragma unroll
    for (int i = 0; i < 4; ++i)
#pragma unroll
      for (int j = 0; j < 2; ++j)
        acc[i][j] = wmma_bf16(af[i], bfr[j], acc[i][j]);
  }
#else
  for (int kt = 0; kt < ntiles; ++kt) {
    const int k0 = kt * 32;
    __syncthreads();
    *(v16s*)&As[0][ldr * 32 + ldc] = *(const v16s*)(gA + k0);
    *(v16s*)&Bs[0][ldr * 32 + ldc] = *(const v16s*)(gB + k0);
    if (kt + 1 < ntiles) {
      __builtin_prefetch(gA + k0 + 32, 0, 1);
      __builtin_prefetch(gB + k0 + 32, 0, 1);
    }
    __syncthreads();
    v16s af[4], bfr[2];
#pragma unroll
    for (int i = 0; i < 4; ++i)
      af[i] = load_a_frag(&As[0][(wm * 64 + i * 16 + l16) * 32], hf);
#pragma unroll
    for (int j = 0; j < 2; ++j)
      bfr[j] = load_b_frag(&Bs[0][(wn * 32 + j * 16 + l16) * 32], hf);
#pragma unroll
    for (int i = 0; i < 4; ++i)
#pragma unroll
      for (int j = 0; j < 2; ++j)
        acc[i][j] = wmma_bf16(af[i], bfr[j], acc[i][j]);
  }
#endif

#pragma unroll
  for (int i = 0; i < 4; ++i)
#pragma unroll
    for (int j = 0; j < 2; ++j)
#pragma unroll
      for (int v = 0; v < 8; ++v) {
        int row = bm + wm * 64 + i * 16 + hf * 8 + v;
        int col = bn + wn * 32 + j * 16 + l16;
        C[(size_t)row * N + col] = acc[i][j][v];
      }
}

// RoPE on Q/K + repack Q,K,V from [b*s, h*dk] f32 to head-major [b,h,s,dk] bf16
__global__ __launch_bounds__(256)
void rope_pack(const float* __restrict__ Qf, const float* __restrict__ Kf,
               const float* __restrict__ Vf, const int* __restrict__ pos,
               unsigned short* __restrict__ Qh, unsigned short* __restrict__ Kh,
               unsigned short* __restrict__ Vh) {
  int tid = blockIdx.x * 256 + threadIdx.x;   // [0, 4096*512)
  int row = tid >> 9;                         // b*2048 + s
  int p   = tid & 511;
  int h   = p >> 5;                           // head
  int t2  = p & 31;                           // pair index within dk/2
  size_t src = (size_t)row * 1024 + h * 64 + 2 * t2;
  float pv = (float)pos[row];
  // inv_freq = theta^(-2*t2/64) = exp(-t2 * ln(1e4)/32)
  float freq = __expf(-(float)t2 * 0.2878231366242557f);
  float sn, cs;
  __sincosf(pv * freq, &sn, &cs);
  size_t dst = ((size_t)((row >> 11) * 16 + h) * 2048 + (size_t)(row & 2047)) * 64 + 2 * t2;
  {
    float x1 = Qf[src], x2 = Qf[src + 1];
    Qh[dst]     = f2bf(x1 * cs - x2 * sn);
    Qh[dst + 1] = f2bf(x1 * sn + x2 * cs);
  }
  {
    float x1 = Kf[src], x2 = Kf[src + 1];
    Kh[dst]     = f2bf(x1 * cs - x2 * sn);
    Kh[dst + 1] = f2bf(x1 * sn + x2 * cs);
  }
  Vh[dst]     = f2bf(Vf[src]);
  Vh[dst + 1] = f2bf(Vf[src + 1]);
}

// Causal flash attention, one (b*h, 64-row tile) per block; 4 waves x 16 rows.
// Writes O to AO laid out [b, s, h, dk] (== [b,s,d] row-major).
__global__ __launch_bounds__(128)
void flash_attn(const unsigned short* __restrict__ Qh,
                const unsigned short* __restrict__ Kh,
                const unsigned short* __restrict__ Vh,
                float* __restrict__ AO) {
  __shared__ alignas(32) unsigned short Vt[64 * 64];     // V^T: [dk][key]
  __shared__ alignas(32) unsigned short Pl[4][16 * 64];  // per-wave P tile
  const int bh = blockIdx.y;           // b*16 + h
  const int tile = blockIdx.x;         // 64-row tile index
  const int b = bh >> 4, h = bh & 15;
  const size_t hoff = (size_t)bh * 2048 * 64;
  const unsigned short* Q  = Qh + hoff;
  const unsigned short* Kp = Kh + hoff;
  const unsigned short* Vp = Vh + hoff;
  const int tid = threadIdx.x, lane = tid & 31, w = tid >> 5;
  const int l16 = lane & 15, hf = lane >> 4;
  const int r0 = tile * 64 + w * 16;
  const v8f vz = {0, 0, 0, 0, 0, 0, 0, 0};

  v16s qf[2];
#pragma unroll
  for (int c = 0; c < 2; ++c)
    qf[c] = load_a_frag(Q + (size_t)(r0 + l16) * 64 + c * 32, hf);

  v8f o[4];
  float mr[8], lrow[8];
#pragma unroll
  for (int nt = 0; nt < 4; ++nt) o[nt] = vz;
#pragma unroll
  for (int v = 0; v < 8; ++v) { mr[v] = -1e30f; lrow[v] = 0.f; }

  for (int jb = 0; jb <= tile; ++jb) {
    const int kb = jb * 64;
    __syncthreads();   // protect Vt reuse
    {  // stage V tile transposed into LDS
      int vr = tid >> 1, vc0 = (tid & 1) * 32;
      const unsigned short* vs = &Vp[(size_t)(kb + vr) * 64 + vc0];
      v16s v0 = *(const v16s*)vs;
      v16s v1 = *(const v16s*)(vs + 16);
#pragma unroll
      for (int e = 0; e < 16; ++e) {
        Vt[(vc0 + e) * 64 + vr]      = (unsigned short)v0[e];
        Vt[(vc0 + 16 + e) * 64 + vr] = (unsigned short)v1[e];
      }
    }
    __syncthreads();

    float sreg[4][8];
#pragma unroll
    for (int n = 0; n < 4; ++n) {   // S = Q * K^T  (16x16 tile n)
      const unsigned short* krow = Kp + (size_t)(kb + n * 16 + l16) * 64;
      v16s kf0 = load_b_frag(krow, hf);
      v16s kf1 = load_b_frag(krow + 32, hf);
      v8f s = vz;
      s = wmma_bf16(qf[0], kf0, s);
      s = wmma_bf16(qf[1], kf1, s);
#pragma unroll
      for (int v = 0; v < 8; ++v) {
        int rg = r0 + hf * 8 + v;
        int cg = kb + n * 16 + l16;
        sreg[n][v] = (cg <= rg) ? s[v] * 0.125f : -1e30f;  // 1/sqrt(64), causal
      }
    }
    float mn[8], al[8], rs[8];
#pragma unroll
    for (int v = 0; v < 8; ++v) {   // row max across 64 cols (regs + 16 lanes)
      float m = fmaxf(fmaxf(sreg[0][v], sreg[1][v]), fmaxf(sreg[2][v], sreg[3][v]));
      m = fmaxf(m, __shfl_xor(m, 1, 32));
      m = fmaxf(m, __shfl_xor(m, 2, 32));
      m = fmaxf(m, __shfl_xor(m, 4, 32));
      m = fmaxf(m, __shfl_xor(m, 8, 32));
      mn[v] = fmaxf(mr[v], m);
      al[v] = __expf(mr[v] - mn[v]);
      mr[v] = mn[v];
      rs[v] = 0.f;
    }
#pragma unroll
    for (int n = 0; n < 4; ++n)
#pragma unroll
      for (int v = 0; v < 8; ++v) {
        float pp = __expf(sreg[n][v] - mn[v]);
        sreg[n][v] = pp;
        rs[v] += pp;
      }
#pragma unroll
    for (int v = 0; v < 8; ++v) {
      float r = rs[v];
      r += __shfl_xor(r, 1, 32);
      r += __shfl_xor(r, 2, 32);
      r += __shfl_xor(r, 4, 32);
      r += __shfl_xor(r, 8, 32);
      lrow[v] = lrow[v] * al[v] + r;
    }
#pragma unroll
    for (int nt = 0; nt < 4; ++nt)
#pragma unroll
      for (int v = 0; v < 8; ++v) o[nt][v] = o[nt][v] * al[v];

    // D-layout P -> LDS -> A-layout fragments
#pragma unroll
    for (int n = 0; n < 4; ++n)
#pragma unroll
      for (int v = 0; v < 8; ++v)
        Pl[w][(hf * 8 + v) * 64 + n * 16 + l16] = f2bf(sreg[n][v]);
    __syncthreads();
    v16s pf0 = load_a_frag(&Pl[w][l16 * 64], hf);
    v16s pf1 = load_a_frag(&Pl[w][l16 * 64 + 32], hf);
#pragma unroll
    for (int nt = 0; nt < 4; ++nt) {   // O += P * V
      v16s vf0 = load_b_frag(&Vt[(nt * 16 + l16) * 64], hf);
      v16s vf1 = load_b_frag(&Vt[(nt * 16 + l16) * 64 + 32], hf);
      o[nt] = wmma_bf16(pf0, vf0, o[nt]);
      o[nt] = wmma_bf16(pf1, vf1, o[nt]);
    }
  }
  const size_t rowbase = (size_t)b * 2048;
#pragma unroll
  for (int nt = 0; nt < 4; ++nt)
#pragma unroll
    for (int v = 0; v < 8; ++v) {
      int rg = r0 + hf * 8 + v;
      AO[(rowbase + rg) * 1024 + h * 64 + nt * 16 + l16] = o[nt][v] / lrow[v];
    }
}

extern "C" void kernel_launch(void* const* d_in, const int* in_sizes, int n_in,
                              void* d_out, int out_size, void* d_ws, size_t ws_size,
                              hipStream_t stream) {
  (void)in_sizes; (void)n_in; (void)out_size; (void)ws_size;
  const float* x  = (const float*)d_in[0];
  const int*   pos = (const int*)d_in[1];
  const float* Wq = (const float*)d_in[2];
  const float* Wk = (const float*)d_in[3];
  const float* Wv = (const float*)d_in[4];
  const float* Wo = (const float*)d_in[5];
  float* out = (float*)d_out;
  char* ws = (char*)d_ws;

  const size_t MB = 1ull << 20;
  unsigned short* Xb  = (unsigned short*)(ws + 0);        //  8 MB: x bf16
  unsigned short* Wqb = (unsigned short*)(ws + 8  * MB);  //  2 MB each
  unsigned short* Wkb = (unsigned short*)(ws + 10 * MB);
  unsigned short* Wvb = (unsigned short*)(ws + 12 * MB);
  unsigned short* Wob = (unsigned short*)(ws + 14 * MB);
  float* Qf = (float*)(ws + 16 * MB);                     // 16 MB each (f32 proj)
  float* Kf = (float*)(ws + 32 * MB);
  float* Vf = (float*)(ws + 48 * MB);
  unsigned short* Qh2 = (unsigned short*)(ws + 64 * MB);  //  8 MB each (head-major bf16)
  unsigned short* Kh2 = (unsigned short*)(ws + 72 * MB);
  unsigned short* Vh2 = (unsigned short*)(ws + 80 * MB);  // ends at 88 MB
  float* AO = Qf;                                         // reuse Qf region (16 MB)
  unsigned short* AOb = (unsigned short*)Kf;              // reuse Kf region (8 MB)

  const int M = 4096, D = 1024;

  cvt_f32_bf16<<<(M * D) / 256, 256, 0, stream>>>(x, Xb, M * D);
  cvt_f32_bf16<<<(D * D) / 256, 256, 0, stream>>>(Wq, Wqb, D * D);
  cvt_f32_bf16<<<(D * D) / 256, 256, 0, stream>>>(Wk, Wkb, D * D);
  cvt_f32_bf16<<<(D * D) / 256, 256, 0, stream>>>(Wv, Wvb, D * D);
  cvt_f32_bf16<<<(D * D) / 256, 256, 0, stream>>>(Wo, Wob, D * D);

  dim3 gg(D / 128, M / 128);
  gemm_nt_bf16<<<gg, 256, 0, stream>>>(Xb, Wqb, Qf, M, D, D);
  gemm_nt_bf16<<<gg, 256, 0, stream>>>(Xb, Wkb, Kf, M, D, D);
  gemm_nt_bf16<<<gg, 256, 0, stream>>>(Xb, Wvb, Vf, M, D, D);

  rope_pack<<<(M * 512) / 256, 256, 0, stream>>>(Qf, Kf, Vf, pos, Qh2, Kh2, Vh2);

  flash_attn<<<dim3(32, 32), 128, 0, stream>>>(Qh2, Kh2, Vh2, AO);

  cvt_f32_bf16<<<(M * D) / 256, 256, 0, stream>>>(AO, AOb, M * D);
  gemm_nt_bf16<<<gg, 256, 0, stream>>>(AOb, Wob, out, M, D, D);
}
